// Agent_45535243272418
// MI455X (gfx1250) — matrix operand order repaired and checked by reading, not compiled, not run on hardware
//
#include <hip/hip_runtime.h>
#include <hip/hip_bf16.h>
#include <math.h>

typedef __attribute__((ext_vector_type(16))) _Float16 v16h;
typedef __attribute__((ext_vector_type(8)))  float    v8f;

// ---------------- problem constants ----------------
#define BB   64
#define CIN  4
#define HH   84
#define HID  512
#define MEM  64
#define LSEQ 65          // MEM + 1
#define DST  16
#define DCONV 4
#define DIN  1024        // EXP*HID
#define DTR  32          // HID/16
#define NACT 6

// conv geometry
#define C1O 32
#define H1  20
#define C2O 64
#define H2  9
#define C3O 64
#define H3  7
#define FLAT (C3O*H3*H3)   // 3136

// ================= generic conv2d + ReLU =================
__global__ void conv2d_relu_k(const float* __restrict__ in, const float* __restrict__ wt,
                              const float* __restrict__ bias, float* __restrict__ out,
                              int Bn, int Ci, int Hi, int Wi,
                              int Co, int Kk, int S, int Ho, int Wo, float inScale)
{
    int i = blockIdx.x * blockDim.x + threadIdx.x;
    int total = Bn * Co * Ho * Wo;
    if (i >= total) return;
    int ox = i % Wo; int t = i / Wo;
    int oy = t % Ho; t /= Ho;
    int oc = t % Co; int bb = t / Co;

    float acc = bias[oc];
    const float* wp = wt + (size_t)oc * Ci * Kk * Kk;
    const float* ip = in + (size_t)bb * Ci * Hi * Wi;
    for (int ic = 0; ic < Ci; ++ic) {
        for (int ky = 0; ky < Kk; ++ky) {
            const float* irow = ip + ((size_t)ic * Hi + (oy * S + ky)) * Wi + ox * S;
            const float* wrow = wp + ((size_t)ic * Kk + ky) * Kk;
            for (int kx = 0; kx < Kk; ++kx)
                acc += irow[kx] * inScale * wrow[kx];
        }
    }
    out[i] = fmaxf(acc, 0.0f);
}

// ================= WMMA f16 GEMM: C = act(A * W^T + bias) =================
// One wave computes a 32x64 output tile: MT=2 x NT=4 accumulators,
// 8 independent v_wmma_f32_16x16x32_f16 per K-step (no RAW chain between them).
// A: (M x K) row-major, stride lda (f32);  W: (N x K) row-major, stride ldw
// C: (M x N) row-major, stride ldc.
// Requires: M % 32 == 0, N % 64 == 0, K % 32 == 0.
// ACT: 0=none, 1=relu, 3=softplus
#define MT 2
#define NT 4
template<int ACT, bool HAS_BIAS>
__global__ __launch_bounds__(256)
void gemm_wmma_k(const float* __restrict__ A, int lda,
                 const float* __restrict__ W, int ldw,
                 const float* __restrict__ bias,
                 float* __restrict__ C, int ldc,
                 int M, int N, int K)
{
    const int lane = threadIdx.x;                 // 0..31, one wave per 32x64 tile
    const int ntN  = N >> 6;                      // tiles of 64 in N
    const int ntM  = M >> 5;                      // tiles of 32 in M
    const int tile = blockIdx.x * blockDim.y + threadIdx.y;
    if (tile >= ntM * ntN) return;                // wave-uniform exit (EXEC all-1s for WMMA)
    const int tm = tile / ntN;
    const int tn = tile % ntN;
    const int ml = lane & 15;
    const int hi = lane >> 4;

    const float* arow[MT];
    #pragma unroll
    for (int mt = 0; mt < MT; ++mt)
        arow[mt] = A + (size_t)(tm * 32 + mt * 16 + ml) * lda;   // A row m
    const float* brow[NT];
    #pragma unroll
    for (int nt = 0; nt < NT; ++nt)
        brow[nt] = W + (size_t)(tn * 64 + nt * 16 + ml) * ldw;   // B col n == W row n

    v8f acc[MT][NT];
    #pragma unroll
    for (int mt = 0; mt < MT; ++mt)
        #pragma unroll
        for (int nt = 0; nt < NT; ++nt)
            acc[mt][nt] = (v8f){};

    for (int kk = 0; kk < K; kk += 32) {
        v16h av[MT], bv[NT];
        // A 16x32 f16 layout: lane(hi) K-ranges [hi*8, hi*8+8) and [16+hi*8, ...)
        #pragma unroll
        for (int mt = 0; mt < MT; ++mt) {
            const float* ap = arow[mt] + kk + hi * 8;
            __builtin_prefetch(ap + 32, 0, 1);           // next K-step line
            #pragma unroll
            for (int e = 0; e < 8; ++e)  av[mt][e]     = (_Float16)ap[e];
            #pragma unroll
            for (int e = 0; e < 8; ++e)  av[mt][8 + e] = (_Float16)ap[16 + e];
        }
        // B 32x16 f16 layout: lane(hi) K-ranges [hi*8, hi*8+8) and [16+hi*8, ...)
        #pragma unroll
        for (int nt = 0; nt < NT; ++nt) {
            const float* bp = brow[nt] + kk + hi * 8;
            __builtin_prefetch(bp + 32, 0, 1);
            #pragma unroll
            for (int e = 0; e < 8; ++e)  bv[nt][e]     = (_Float16)bp[e];
            #pragma unroll
            for (int e = 0; e < 8; ++e)  bv[nt][8 + e] = (_Float16)bp[16 + e];
        }
        #pragma unroll
        for (int mt = 0; mt < MT; ++mt)
            #pragma unroll
            for (int nt = 0; nt < NT; ++nt)
                acc[mt][nt] = __builtin_amdgcn_wmma_f32_16x16x32_f16(
                    false, av[mt], false, bv[nt], (short)0, acc[mt][nt], false, false);
    }

    #pragma unroll
    for (int nt = 0; nt < NT; ++nt) {
        const int n = tn * 64 + nt * 16 + ml;
        const float bval = HAS_BIAS ? bias[n] : 0.0f;
        #pragma unroll
        for (int mt = 0; mt < MT; ++mt) {
            #pragma unroll
            for (int r = 0; r < 8; ++r) {
                int m = tm * 32 + mt * 16 + hi * 8 + r;   // D: m = r + 8*hi, n = lane&15
                float v = acc[mt][nt][r] + bval;
                if (ACT == 1) v = fmaxf(v, 0.0f);
                else if (ACT == 3) v = (v > 20.0f) ? v : log1pf(expf(v));   // softplus
                C[(size_t)m * ldc + n] = v;
            }
        }
    }
}

// ================= build seq = concat(memory, encoded) =================
__global__ void build_seq_k(const float* __restrict__ mem, const float* __restrict__ enc,
                            float* __restrict__ seq)
{
    int i = blockIdx.x * blockDim.x + threadIdx.x;
    if (i >= BB * LSEQ * HID) return;
    int d = i & (HID - 1);
    int l = (i >> 9) % LSEQ;
    int b = i / (LSEQ * HID);
    seq[i] = (l < MEM) ? mem[((size_t)b * MEM + l) * HID + d] : enc[(size_t)b * HID + d];
}

// ================= depthwise causal conv1d + SiLU =================
__global__ void dwconv_silu_k(const float* __restrict__ xm, const float* __restrict__ w,
                              const float* __restrict__ bias, float* __restrict__ u)
{
    int i = blockIdx.x * blockDim.x + threadIdx.x;
    if (i >= BB * LSEQ * DIN) return;
    int d = i & (DIN - 1);
    int l = (i >> 10) % LSEQ;
    int b = i / (LSEQ * DIN);
    float acc = bias[d];
    #pragma unroll
    for (int j = 0; j < DCONV; ++j) {
        int ls = l - (DCONV - 1) + j;
        if (ls >= 0)
            acc += xm[((size_t)b * LSEQ + ls) * DIN + d] * w[d * DCONV + j];
    }
    u[i] = acc / (1.0f + expf(-acc));    // silu
}

// ================= selective scan, keep only l = L-1, fused epilogue ========
__global__ void scan_k(const float* __restrict__ delta, const float* __restrict__ u,
                       const float* __restrict__ xdbl, const float* __restrict__ A_log,
                       const float* __restrict__ Dp,   const float* __restrict__ zlast,
                       float* __restrict__ ylast)
{
    int i = blockIdx.x * blockDim.x + threadIdx.x;
    if (i >= BB * DIN) return;
    int d = i & (DIN - 1);
    int b = i >> 10;

    float a[DST];
    #pragma unroll
    for (int n = 0; n < DST; ++n) a[n] = -expf(A_log[(size_t)d * DST + n]);

    float h[DST];
    #pragma unroll
    for (int n = 0; n < DST; ++n) h[n] = 0.0f;

    float yv = 0.0f;
    for (int l = 0; l < LSEQ; ++l) {
        size_t row = (size_t)b * LSEQ + l;
        float dl = delta[row * DIN + d];
        float ul = u[row * DIN + d];
        float du = dl * ul;
        const float* Bm = xdbl + row * (DTR + 2 * DST) + DTR;
        const float* Cm = Bm + DST;
        yv = 0.0f;
        #pragma unroll
        for (int n = 0; n < DST; ++n) {
            h[n] = expf(dl * a[n]) * h[n] + du * Bm[n];
            yv += h[n] * Cm[n];
        }
    }
    // epilogue at final step: (y + u*Dp) * silu(z)
    size_t row = (size_t)b * LSEQ + (LSEQ - 1);
    float ul = u[row * DIN + d];
    float zv = zlast[(size_t)b * DIN + d];
    float sz = zv / (1.0f + expf(-zv));
    ylast[(size_t)b * DIN + d] = (yv + ul * Dp[d]) * sz;
}

// ================= actor/critic heads: one wave per batch element ===========
__global__ void head_k(const float* __restrict__ hid,
                       const float* __restrict__ aw, const float* __restrict__ ab,
                       const float* __restrict__ cw, const float* __restrict__ cb,
                       const int* __restrict__ action, float* __restrict__ out)
{
    int b = blockIdx.x;
    int lane = threadIdx.x;     // 32 threads
    float part[NACT + 1];
    #pragma unroll
    for (int j = 0; j <= NACT; ++j) part[j] = 0.0f;

    for (int d = lane; d < HID; d += 32) {
        float hv = hid[(size_t)b * HID + d];
        #pragma unroll
        for (int aidx = 0; aidx < NACT; ++aidx)
            part[aidx] += hv * aw[(size_t)aidx * HID + d];
        part[NACT] += hv * cw[d];
    }
    #pragma unroll
    for (int off = 16; off > 0; off >>= 1) {
        #pragma unroll
        for (int j = 0; j <= NACT; ++j)
            part[j] += __shfl_down(part[j], off);
    }
    if (lane == 0) {
        float logits[NACT];
        float mx = -1e30f;
        #pragma unroll
        for (int aidx = 0; aidx < NACT; ++aidx) {
            logits[aidx] = part[aidx] + ab[aidx];
            mx = fmaxf(mx, logits[aidx]);
        }
        float se = 0.0f;
        #pragma unroll
        for (int aidx = 0; aidx < NACT; ++aidx) se += expf(logits[aidx] - mx);
        float lse = mx + logf(se);
        float ent = 0.0f;
        #pragma unroll
        for (int aidx = 0; aidx < NACT; ++aidx) {
            float lp = logits[aidx] - lse;
            ent -= expf(lp) * lp;
        }
        int act = action[b];
        out[0 * BB + b] = (float)act;                 // action
        out[1 * BB + b] = logits[act] - lse;          // logp
        out[2 * BB + b] = ent;                        // entropy
        out[3 * BB + b] = part[NACT] + cb[0];         // value
    }
}

// =============================== launch ====================================
extern "C" void kernel_launch(void* const* d_in, const int* in_sizes, int n_in,
                              void* d_out, int out_size, void* d_ws, size_t ws_size,
                              hipStream_t stream)
{
    const float* x        = (const float*)d_in[0];
    const float* memory   = (const float*)d_in[1];
    const int*   action   = (const int*)  d_in[2];
    const float* c1w      = (const float*)d_in[3];
    const float* c1b      = (const float*)d_in[4];
    const float* c2w      = (const float*)d_in[5];
    const float* c2b      = (const float*)d_in[6];
    const float* c3w      = (const float*)d_in[7];
    const float* c3b      = (const float*)d_in[8];
    const float* fcw      = (const float*)d_in[9];
    const float* fcb      = (const float*)d_in[10];
    const float* in_proj_w= (const float*)d_in[11];
    const float* conv1dw  = (const float*)d_in[12];
    const float* conv1db  = (const float*)d_in[13];
    const float* x_proj_w = (const float*)d_in[14];
    const float* dt_proj_w= (const float*)d_in[15];
    const float* dt_proj_b= (const float*)d_in[16];
    const float* A_log    = (const float*)d_in[17];
    const float* Dp       = (const float*)d_in[18];
    const float* out_proj_w=(const float*)d_in[19];
    const float* actor_w  = (const float*)d_in[20];
    const float* actor_b  = (const float*)d_in[21];
    const float* critic_w = (const float*)d_in[22];
    const float* critic_b = (const float*)d_in[23];

    float* out = (float*)d_out;
    float* enc = out + 4 * BB;                  // encoded region of d_out (64x512)

    // ---- workspace layout (floats) ----
    float* ws   = (float*)d_ws;
    float* h1   = ws;                                   // 64*32*20*20 = 819200
    float* h2   = h1 + (size_t)BB * C1O * H1 * H1;      // 64*64*9*9   = 331776
    float* h3   = h2 + (size_t)BB * C2O * H2 * H2;      // 64*3136     = 200704
    float* seq  = h3 + (size_t)BB * FLAT;               // 64*65*512   = 2129920
    float* xm   = seq + (size_t)BB * LSEQ * HID;        // 64*65*1024  = 4259840
    float* u    = xm  + (size_t)BB * LSEQ * DIN;        // 64*65*1024  = 4259840
    float* xdbl = u   + (size_t)BB * LSEQ * DIN;        // 64*65*64    = 266240
    float* delta= xdbl+ (size_t)BB * LSEQ * (DTR + 2*DST); // 64*65*1024
    float* zlast= delta+(size_t)BB * LSEQ * DIN;        // 64*1024
    float* ylast= zlast+(size_t)BB * DIN;               // 64*1024
    float* olast= ylast+(size_t)BB * DIN;               // 64*512

    const int T = 256;
    // ---- CNN encoder ----
    {
        int tot = BB * C1O * H1 * H1;
        conv2d_relu_k<<<(tot + T - 1) / T, T, 0, stream>>>(
            x, c1w, c1b, h1, BB, CIN, HH, HH, C1O, 8, 4, H1, H1, 1.0f / 255.0f);
    }
    {
        int tot = BB * C2O * H2 * H2;
        conv2d_relu_k<<<(tot + T - 1) / T, T, 0, stream>>>(
            h1, c2w, c2b, h2, BB, C1O, H1, H1, C2O, 4, 2, H2, H2, 1.0f);
    }
    {
        int tot = BB * C3O * H3 * H3;
        conv2d_relu_k<<<(tot + T - 1) / T, T, 0, stream>>>(
            h2, c3w, c3b, h3, BB, C2O, H2, H2, C3O, 3, 1, H3, H3, 1.0f);
    }
    // fc: encoded = relu(h3 @ fcw^T + fcb)  -> writes straight into d_out
    {
        dim3 blk(32, 8);
        int tiles = (BB / 32) * (HID / 64);
        gemm_wmma_k<1, true><<<(tiles + 7) / 8, blk, 0, stream>>>(
            h3, FLAT, fcw, FLAT, fcb, enc, HID, BB, HID, FLAT);
    }
    // ---- Mamba block ----
    {
        int tot = BB * LSEQ * HID;
        build_seq_k<<<(tot + T - 1) / T, T, 0, stream>>>(memory, enc, seq);
    }
    // xm half of in_proj for all rows (z half only needed at last step)
    {
        dim3 blk(32, 8);
        int M = BB * LSEQ, N = DIN;
        int tiles = (M / 32) * (N / 64);
        gemm_wmma_k<0, false><<<(tiles + 7) / 8, blk, 0, stream>>>(
            seq, HID, in_proj_w, HID, nullptr, xm, DIN, M, N, HID);
    }
    // z at last step: seq[b,64,:] == encoded[b,:]
    {
        dim3 blk(32, 8);
        int tiles = (BB / 32) * (DIN / 64);
        gemm_wmma_k<0, false><<<(tiles + 7) / 8, blk, 0, stream>>>(
            enc, HID, in_proj_w + (size_t)DIN * HID, HID, nullptr, zlast, DIN, BB, DIN, HID);
    }
    // depthwise causal conv1d + SiLU -> u
    {
        int tot = BB * LSEQ * DIN;
        dwconv_silu_k<<<(tot + T - 1) / T, T, 0, stream>>>(xm, conv1dw, conv1db, u);
    }
    // x_proj: x_dbl = u @ x_proj_w^T   (N = 32+16+16 = 64)
    {
        dim3 blk(32, 8);
        int M = BB * LSEQ, N = DTR + 2 * DST;
        int tiles = (M / 32) * (N / 64);
        gemm_wmma_k<0, false><<<(tiles + 7) / 8, blk, 0, stream>>>(
            u, DIN, x_proj_w, DIN, nullptr, xdbl, N, M, N, DIN);
    }
    // dt_proj + softplus: delta = softplus(x_dbl[...,:32] @ dt_proj_w^T + b)
    {
        dim3 blk(32, 8);
        int M = BB * LSEQ, N = DIN;
        int tiles = (M / 32) * (N / 64);
        gemm_wmma_k<3, true><<<(tiles + 7) / 8, blk, 0, stream>>>(
            xdbl, DTR + 2 * DST, dt_proj_w, DTR, dt_proj_b, delta, DIN, M, N, DTR);
    }
    // selective scan (keep last step, fused gate epilogue)
    {
        int tot = BB * DIN;
        scan_k<<<(tot + T - 1) / T, T, 0, stream>>>(delta, u, xdbl, A_log, Dp, zlast, ylast);
    }
    // out_proj on last step only: hidden = ylast @ out_proj_w^T
    {
        dim3 blk(32, 8);
        int tiles = (BB / 32) * (HID / 64);
        gemm_wmma_k<0, false><<<(tiles + 7) / 8, blk, 0, stream>>>(
            ylast, DIN, out_proj_w, DIN, nullptr, olast, HID, BB, HID, DIN);
    }
    // heads
    head_k<<<BB, 32, 0, stream>>>(olast, actor_w, actor_b, critic_w, critic_b, action, out);
}
